// RelationshipAttention_4913442586853
// MI455X (gfx1250) — matrix operand order repaired and checked by reading, not compiled, not run on hardware
//
#include <hip/hip_runtime.h>
#include <math.h>
#include <stdint.h>

typedef __attribute__((ext_vector_type(16))) __bf16 v16bf;
typedef __attribute__((ext_vector_type(8)))  __bf16 v8bf;
typedef __attribute__((ext_vector_type(8)))  float  v8f;

#define B_  8
#define N_  2048
#define D_  256
#define KI  512   // TOP_K_INST
#define KR  32    // TOP_K_REL

union V16U { v16bf v; v8bf h[2]; };

// allocation-relative LDS address: low 32 bits of a generic pointer to LDS
__device__ __forceinline__ unsigned lds_off(const void* p) {
    return (unsigned)(unsigned long long)p;
}

// ---------------- kernel 0: f32 -> bf16 convert ----------------
__global__ __launch_bounds__(256)
void cvt_bf16(const float* __restrict__ src, unsigned short* __restrict__ dst, int n) {
    int i = blockIdx.x * blockDim.x + threadIdx.x;
    int stride = gridDim.x * blockDim.x;
    for (; i < n; i += stride) {
        __bf16 h = (__bf16)src[i];
        dst[i] = *(const unsigned short*)&h;
    }
}

// ---------------- kernel 1: scores = q @ k^T (bf16 WMMA, f32 acc) ----------------
// workgroup: 128x128 tile, 8 waves; wave (wm,wn) owns 32x64 patch = 2x4 16x16 frags
// LDS panels staged with GLOBAL_LOAD_ASYNC_TO_LDS_B128 (ASYNCcnt path, no staging VGPRs)
__global__ __launch_bounds__(256)
void gemm_scores(const unsigned short* __restrict__ qh_u,
                 const unsigned short* __restrict__ kh_u,
                 float* __restrict__ scores) {
    __shared__ __bf16 As[128 * 64];
    __shared__ __bf16 Bs[128 * 64];

    const int b   = blockIdx.z;
    const int mB  = blockIdx.y * 128;
    const int nB  = blockIdx.x * 128;
    const int tid = threadIdx.x;
    const int wave = tid >> 5, lane = tid & 31;
    const int l = lane & 15, hi = lane >> 4;
    const int wm = wave >> 1, wn = wave & 1;

    v8f acc[2][4];
    for (int mi = 0; mi < 2; ++mi)
        for (int ni = 0; ni < 4; ++ni)
            for (int r = 0; r < 8; ++r) acc[mi][ni][r] = 0.f;

    const unsigned short* qb = qh_u + ((size_t)b * N_ + mB) * D_;
    const unsigned short* kb = kh_u + ((size_t)b * N_ + nB) * D_;

    for (int ks = 0; ks < D_; ks += 64) {
        // stage 128x64 bf16 panels: 1024 16B chunks each, 4 per thread, async to LDS
        for (int it = 0; it < 4; ++it) {
            int chunk = tid + it * 256;          // 0..1023
            int row = chunk >> 3;
            int co  = (chunk & 7) * 8;
            unsigned ldsA = lds_off(&As[row * 64 + co]);
            unsigned ldsB = lds_off(&Bs[row * 64 + co]);
            unsigned gofs = (unsigned)(((unsigned)row * D_ + (unsigned)(ks + co)) * 2u);
            asm volatile("global_load_async_to_lds_b128 %0, %1, %2 offset:0"
                         :: "v"(ldsA), "v"(gofs), "s"(qb) : "memory");
            asm volatile("global_load_async_to_lds_b128 %0, %1, %2 offset:0"
                         :: "v"(ldsB), "v"(gofs), "s"(kb) : "memory");
        }
        asm volatile("s_wait_asynccnt 0x0" ::: "memory");
        __syncthreads();

        for (int kk = 0; kk < 64; kk += 32) {
            V16U a[2], bb[4];
            for (int mi = 0; mi < 2; ++mi) {
                int row = wm * 32 + mi * 16 + l;
                // A loaded in contiguous k order, same as B: half-wave 0 -> K 0-15, half-wave 1 -> K 16-31 (contiguous 32B)
                a[mi].v = *(const v16bf*)&As[row * 64 + kk + hi * 16];
            }
            for (int ni = 0; ni < 4; ++ni) {
                int col = wn * 64 + ni * 16 + l;
                // B loaded in contiguous k order, same as A: lane<16 -> K 0-15, lane>=16 -> K 16-31 (contiguous 32B)
                bb[ni].v = *(const v16bf*)&Bs[col * 64 + kk + hi * 16];
            }
            for (int mi = 0; mi < 2; ++mi)
                for (int ni = 0; ni < 4; ++ni)
                    acc[mi][ni] = __builtin_amdgcn_wmma_f32_16x16x32_bf16(
                        false, a[mi].v, false, bb[ni].v,
                        (short)0, acc[mi][ni], false, false);
        }
        __syncthreads();
    }

    // C/D layout: VGPR r -> M = hi*8 + r, N = l
    for (int mi = 0; mi < 2; ++mi)
        for (int ni = 0; ni < 4; ++ni) {
            int col = nB + wn * 64 + ni * 16 + l;
            int rb  = mB + wm * 32 + mi * 16 + hi * 8;
            for (int r = 0; r < 8; ++r)
                scores[((size_t)b * N_ + rb + r) * N_ + col] = acc[mi][ni][r];
        }
}

// ---------------- kernel 2: per-row softmax stats -> diagonal prob ----------------
__global__ __launch_bounds__(256)
void row_softmax_diag(const float* __restrict__ scores, float* __restrict__ diag) {
    __shared__ float red[256];
    int bi = blockIdx.x;                  // 0 .. B*N-1
    int b = bi >> 11, i = bi & (N_ - 1);
    const float* row = scores + ((size_t)b * N_ + i) * N_;
    int tid = threadIdx.x;

    float m = -3.4e38f;
    for (int j = tid; j < N_; j += 256) m = fmaxf(m, row[j]);
    red[tid] = m; __syncthreads();
    for (int s = 128; s > 0; s >>= 1) {
        if (tid < s) red[tid] = fmaxf(red[tid], red[tid + s]);
        __syncthreads();
    }
    float rmax = red[0]; __syncthreads();

    float sum = 0.f;
    for (int j = tid; j < N_; j += 256) sum += __expf(row[j] - rmax);
    red[tid] = sum; __syncthreads();
    for (int s = 128; s > 0; s >>= 1) {
        if (tid < s) red[tid] += red[tid + s];
        __syncthreads();
    }
    if (tid == 0) diag[bi] = __expf(row[i] - rmax) / red[0];
}

// ---------------- kernel 3: per-batch top-512 of diag, ascending indices ----------------
__global__ __launch_bounds__(256)
void topk_inst(const float* __restrict__ diag, int* __restrict__ inst) {
    __shared__ unsigned long long keys[N_];
    __shared__ unsigned int idxs[KI];
    int b = blockIdx.x, tid = threadIdx.x;

    for (int i = tid; i < N_; i += 256) {
        unsigned int bits = __float_as_uint(diag[b * N_ + i]); // probs > 0: bits monotone
        keys[i] = ((unsigned long long)bits << 32) | (unsigned int)i;
    }
    __syncthreads();

    for (int k = 2; k <= N_; k <<= 1)
        for (int j = k >> 1; j > 0; j >>= 1) {
            for (int i = tid; i < N_; i += 256) {
                int ixj = i ^ j;
                if (ixj > i) {
                    bool desc = ((i & k) == 0);
                    unsigned long long x = keys[i], y = keys[ixj];
                    if (desc ? (x < y) : (x > y)) { keys[i] = y; keys[ixj] = x; }
                }
            }
            __syncthreads();
        }

    for (int i = tid; i < KI; i += 256) idxs[i] = (unsigned int)(keys[i] & 0xffffffffu);
    __syncthreads();

    for (int k = 2; k <= KI; k <<= 1)
        for (int j = k >> 1; j > 0; j >>= 1) {
            for (int i = tid; i < KI; i += 256) {
                int ixj = i ^ j;
                if (ixj > i) {
                    bool asc = ((i & k) == 0);
                    unsigned int x = idxs[i], y = idxs[ixj];
                    if (asc ? (x > y) : (x < y)) { idxs[i] = y; idxs[ixj] = x; }
                }
            }
            __syncthreads();
        }

    for (int i = tid; i < KI; i += 256) inst[b * KI + i] = (int)idxs[i];
}

// order-preserving float -> uint key (handles negatives)
__device__ __forceinline__ unsigned int fkey(float f) {
    unsigned int b = __float_as_uint(f);
    return b ^ (unsigned int)(((int)b >> 31) | 0x80000000);
}

// ---------------- kernel 4: per selected row, top-32 relationship cols -> soi ----------------
__global__ __launch_bounds__(256)
void topk_rel(const float* __restrict__ scores, const int* __restrict__ inst,
              int* __restrict__ soi) {
    __shared__ int ii[KI];
    __shared__ unsigned long long keys[KI];
    __shared__ int cols[KR];
    int r = blockIdx.x, b = blockIdx.y, tid = threadIdx.x;

    for (int c = tid; c < KI; c += 256) ii[c] = inst[b * KI + c];
    __syncthreads();
    int irow = ii[r];
    const float* row = scores + ((size_t)b * N_ + irow) * N_;

    // softmax is monotone per-row: raw scores give the same top-k as probs
    for (int c = tid; c < KI; c += 256) {
        float v = (c == r) ? 1e9f : row[ii[c]];
        keys[c] = ((unsigned long long)fkey(v) << 32) | (unsigned int)c;
    }
    __syncthreads();

    for (int k = 2; k <= KI; k <<= 1)
        for (int j = k >> 1; j > 0; j >>= 1) {
            for (int i = tid; i < KI; i += 256) {
                int ixj = i ^ j;
                if (ixj > i) {
                    bool desc = ((i & k) == 0);
                    unsigned long long x = keys[i], y = keys[ixj];
                    if (desc ? (x < y) : (x > y)) { keys[i] = y; keys[ixj] = x; }
                }
            }
            __syncthreads();
        }

    if (tid < KR) cols[tid] = (int)(keys[tid] & 0xffffffffu);
    __syncthreads();
    if (tid == 0) {  // ascending column order (32 elems: serial insertion sort)
        for (int a2 = 1; a2 < KR; ++a2) {
            int v = cols[a2], j2 = a2 - 1;
            while (j2 >= 0 && cols[j2] > v) { cols[j2 + 1] = cols[j2]; --j2; }
            cols[j2 + 1] = v;
        }
    }
    __syncthreads();
    if (tid < KR) {
        size_t p = ((size_t)b * KI + r) * KR + tid;
        soi[p * 3 + 0] = b;
        soi[p * 3 + 1] = irow;
        soi[p * 3 + 2] = ii[cols[tid]];
    }
}

// ---------------- kernel 5: rel_e = layernorm(q[subj] + q[obj]) ----------------
__global__ __launch_bounds__(256)
void rel_embed(const float* __restrict__ q, const int* __restrict__ soi,
               float* __restrict__ rel) {
    __shared__ float red[256];
    int p = blockIdx.x, b = blockIdx.y, d = threadIdx.x;
    size_t pr = (size_t)b * (KI * KR) + p;
    int subj = soi[pr * 3 + 1], obj = soi[pr * 3 + 2];

    float e = q[((size_t)b * N_ + subj) * D_ + d] + q[((size_t)b * N_ + obj) * D_ + d];
    red[d] = e; __syncthreads();
    for (int s = 128; s > 0; s >>= 1) { if (d < s) red[d] += red[d + s]; __syncthreads(); }
    float mu = red[0] * (1.f / D_); __syncthreads();
    float c = e - mu;
    red[d] = c * c; __syncthreads();
    for (int s = 128; s > 0; s >>= 1) { if (d < s) red[d] += red[d + s]; __syncthreads(); }
    float var = red[0] * (1.f / D_);   // biased, matches F.layer_norm
    rel[pr * D_ + d] = c * rsqrtf(var + 1e-5f);
}

// ---------------- host launch ----------------
extern "C" void kernel_launch(void* const* d_in, const int* in_sizes, int n_in,
                              void* d_out, int out_size, void* d_ws, size_t ws_size,
                              hipStream_t stream) {
    const float* q = (const float*)d_in[0];
    const float* k = (const float*)d_in[1];

    // d_out: scores1 [B*N*N] f32, soi [B*KI*KR*3] int32, rel_e [B*KI*KR*D] f32
    float* scores = (float*)d_out;
    int*   soi    = (int*)((float*)d_out + (size_t)B_ * N_ * N_);
    float* rel    = (float*)d_out + (size_t)B_ * N_ * N_ + (size_t)B_ * KI * KR * 3;

    // workspace: qh/kh bf16, diag f32, inst int
    char* ws = (char*)d_ws;
    unsigned short* qh  = (unsigned short*)ws;
    unsigned short* kh  = (unsigned short*)(ws + (size_t)B_ * N_ * D_ * 2);
    float*          dia = (float*)(ws + (size_t)B_ * N_ * D_ * 4);
    int*            ins = (int*)(ws + (size_t)B_ * N_ * D_ * 4 + (size_t)B_ * N_ * 4);

    int nel = B_ * N_ * D_;
    cvt_bf16<<<1024, 256, 0, stream>>>(q, qh, nel);
    cvt_bf16<<<1024, 256, 0, stream>>>(k, kh, nel);
    gemm_scores<<<dim3(N_ / 128, N_ / 128, B_), 256, 0, stream>>>(qh, kh, scores);
    row_softmax_diag<<<B_ * N_, 256, 0, stream>>>(scores, dia);
    topk_inst<<<B_, 256, 0, stream>>>(dia, ins);
    topk_rel<<<dim3(KI, B_), 256, 0, stream>>>(scores, ins, soi);
    rel_embed<<<dim3(KI * KR, B_), 256, 0, stream>>>(q, soi, rel);
}
